// MyHopfield_62620623175968
// MI455X (gfx1250) — matrix operand order repaired and checked by reading, not compiled, not run on hardware
//
#include <hip/hip_runtime.h>
#include <stdint.h>

// Problem constants (match reference)
#define D_DIM   1024
#define H_HEADS 16
#define DH_DIM  64
#define T_SEQ   2048
#define B_BATCH 2
#define M_ROWS  (B_BATCH * T_SEQ)   // 4096

typedef __attribute__((ext_vector_type(16))) __bf16 v16bf;
typedef __attribute__((ext_vector_type(8)))  float  v8f;

union FragB {
    v16bf v;
    uint4 u[2];
    unsigned short s[16];
};

// native bf16 convert (RNE) -- lowers to v_cvt_pk_bf16_f32 on gfx1250
__device__ __forceinline__ unsigned short f2bf(float f) {
    union { __bf16 b; unsigned short s; } c;
    c.b = (__bf16)f;
    return c.s;
}

// raw hardware base-2 exp (no denorm-range guard; our inputs are in [-2,2])
__device__ __forceinline__ float fast_exp2(float x) {
    return __builtin_amdgcn_exp2f(x);
}

// ---------------------------------------------------------------------------
// Kernel 1: fused triple layernorm (shared mean/var), fp32 -> bf16 x3
// ---------------------------------------------------------------------------
__global__ void __launch_bounds__(256)
ln3_kernel(const float* __restrict__ x,
           const float* __restrict__ gq, const float* __restrict__ bq,
           const float* __restrict__ gk, const float* __restrict__ bk,
           const float* __restrict__ gv, const float* __restrict__ bv,
           unsigned short* __restrict__ qn,
           unsigned short* __restrict__ kn,
           unsigned short* __restrict__ vn)
{
    const int row  = blockIdx.x;
    const int tid  = threadIdx.x;
    const int c0   = tid * 4;
    const int lane = tid & 31, wid = tid >> 5;

    const float4 xv = *(const float4*)&x[(size_t)row * D_DIM + c0];
    float s  = xv.x + xv.y + xv.z + xv.w;
    float s2 = xv.x * xv.x + xv.y * xv.y + xv.z * xv.z + xv.w * xv.w;

    #pragma unroll
    for (int off = 16; off > 0; off >>= 1) {
        s  += __shfl_down(s,  off);
        s2 += __shfl_down(s2, off);
    }
    __shared__ float red[16];
    __shared__ float stats[2];
    if (lane == 0) { red[wid] = s; red[8 + wid] = s2; }
    __syncthreads();
    if (tid == 0) {
        float ts = 0.f, ts2 = 0.f;
        #pragma unroll
        for (int i = 0; i < 8; i++) { ts += red[i]; ts2 += red[8 + i]; }
        float mu  = ts * (1.0f / D_DIM);
        float var = ts2 * (1.0f / D_DIM) - mu * mu;
        stats[0] = mu;
        stats[1] = rsqrtf(var + 1e-5f);
    }
    __syncthreads();
    const float mu = stats[0], rs = stats[1];

    float xn[4] = { (xv.x - mu) * rs, (xv.y - mu) * rs,
                    (xv.z - mu) * rs, (xv.w - mu) * rs };

    const size_t base = (size_t)row * D_DIM + c0;
    {
        unsigned short t[4];
        #pragma unroll
        for (int i = 0; i < 4; i++) t[i] = f2bf(xn[i] * gq[c0 + i] + bq[c0 + i]);
        *(uint2*)&qn[base] = *(uint2*)t;
    }
    {
        unsigned short t[4];
        #pragma unroll
        for (int i = 0; i < 4; i++) t[i] = f2bf(xn[i] * gk[c0 + i] + bk[c0 + i]);
        *(uint2*)&kn[base] = *(uint2*)t;
    }
    {
        unsigned short t[4];
        #pragma unroll
        for (int i = 0; i < 4; i++) t[i] = f2bf(xn[i] * gv[c0 + i] + bv[c0 + i]);
        *(uint2*)&vn[base] = *(uint2*)t;
    }
}

// ---------------------------------------------------------------------------
// Kernel 2: fp32 -> bf16 convert (weights)
// ---------------------------------------------------------------------------
__global__ void cvt_kernel(const float* __restrict__ in,
                           unsigned short* __restrict__ out, int n)
{
    for (int i = blockIdx.x * blockDim.x + threadIdx.x; i < n;
         i += gridDim.x * blockDim.x)
        out[i] = f2bf(in[i]);
}

// ---------------------------------------------------------------------------
// Kernel 3: Y[M,N] = A[M,K] @ W[N,K]^T (+bias)(+residual), bf16 WMMA, f32 acc
// Block: 256 thr = 8 waves (2x4), wave tile 32x32 (2x2 WMMA), block 64x128.
// ---------------------------------------------------------------------------
template <int OUT_BF16, int ADD_BIAS, int ADD_RES>
__global__ void __launch_bounds__(256)
gemm_nt(const unsigned short* __restrict__ A,
        const unsigned short* __restrict__ W,
        const float* __restrict__ bias,
        const float* __restrict__ res,
        void* __restrict__ outp,
        int M, int N, int K)
{
    const int tid  = threadIdx.x;
    const int lane = tid & 31, w = tid >> 5;
    const int half = lane >> 4, r = lane & 15;
    const int wm = w >> 2, wn = w & 3;
    const int row0 = blockIdx.x * 64  + wm * 32;
    const int col0 = blockIdx.y * 128 + wn * 32;

    const v8f vzero = {0.f, 0.f, 0.f, 0.f, 0.f, 0.f, 0.f, 0.f};
    v8f acc[2][2];
    #pragma unroll
    for (int i = 0; i < 2; i++)
        #pragma unroll
        for (int j = 0; j < 2; j++) acc[i][j] = vzero;

    for (int k0 = 0; k0 < K; k0 += 32) {
        FragB a[2], b[2];
        #pragma unroll
        for (int i = 0; i < 2; i++) {
            const unsigned short* ap = A + (size_t)(row0 + i * 16 + r) * K + k0;
            a[i].u[0] = *(const uint4*)(ap + half * 16);       // K = h*16 .. +7
            a[i].u[1] = *(const uint4*)(ap + half * 16 + 8);   // K = h*16+8 ..
        }
        #pragma unroll
        for (int j = 0; j < 2; j++) {
            const unsigned short* wp =
                W + (size_t)(col0 + j * 16 + r) * K + k0 + half * 16;
            b[j].u[0] = *(const uint4*)(wp);
            b[j].u[1] = *(const uint4*)(wp + 8);
        }
        if (k0 + 32 < K) {  // prefetch next K-slice (global_prefetch_b8)
            __builtin_prefetch(A + (size_t)(row0 + r) * K + k0 + 32, 0, 1);
            __builtin_prefetch(W + (size_t)(col0 + r) * K + k0 + 32, 0, 1);
        }
        #pragma unroll
        for (int i = 0; i < 2; i++)
            #pragma unroll
            for (int j = 0; j < 2; j++)
                acc[i][j] = __builtin_amdgcn_wmma_f32_16x16x32_bf16(
                    false, a[i].v, false, b[j].v, (short)0, acc[i][j],
                    false, false);
    }

    #pragma unroll
    for (int i = 0; i < 2; i++)
        #pragma unroll
        for (int j = 0; j < 2; j++)
            #pragma unroll
            for (int vv = 0; vv < 8; vv++) {
                const int rr = row0 + i * 16 + vv + 8 * half;
                const int cc = col0 + j * 16 + r;
                float val = acc[i][j][vv];
                if (ADD_BIAS) val += bias[cc];
                if (ADD_RES)  val += res[(size_t)rr * N + cc];
                if (OUT_BF16)
                    ((unsigned short*)outp)[(size_t)rr * N + cc] = f2bf(val);
                else
                    ((float*)outp)[(size_t)rr * N + cc] = val;
            }
}

// ---------------------------------------------------------------------------
// Kernel 4: row-wise L2 norm over D, optional per-head beta/sqrt(DH) fold.
// For q (use_beta=1) we additionally fold log2(e) so attention can use the
// native base-2 exp: 2^(s*log2e) == e^s exactly.
// ---------------------------------------------------------------------------
__global__ void __launch_bounds__(256)
l2norm_kernel(const float* __restrict__ in,
              unsigned short* __restrict__ out,
              const float* __restrict__ beta_param, int use_beta)
{
    const int row  = blockIdx.x;
    const int tid  = threadIdx.x;
    const int c0   = tid * 4;
    const int lane = tid & 31, wid = tid >> 5;

    const float4 xv = *(const float4*)&in[(size_t)row * D_DIM + c0];
    float s2 = xv.x * xv.x + xv.y * xv.y + xv.z * xv.z + xv.w * xv.w;
    #pragma unroll
    for (int off = 16; off > 0; off >>= 1) s2 += __shfl_down(s2, off);

    __shared__ float red[8];
    __shared__ float invn;
    if (lane == 0) red[wid] = s2;
    __syncthreads();
    if (tid == 0) {
        float ts = 0.f;
        #pragma unroll
        for (int i = 0; i < 8; i++) ts += red[i];
        invn = 1.0f / fmaxf(sqrtf(ts), 1e-6f);
    }
    __syncthreads();

    float scl = invn;
    if (use_beta) {
        const int h = c0 >> 6;                              // head index
        const float bp = beta_param[h];
        float sp = (bp > 20.f) ? bp : log1pf(__expf(bp));   // softplus
        sp = fminf(fmaxf(sp, 0.1f), 10.0f);                 // clip
        scl *= sp * 0.125f * 1.44269504088896340736f;       // beta/8 * log2(e)
    }

    unsigned short t[4] = { f2bf(xv.x * scl), f2bf(xv.y * scl),
                            f2bf(xv.z * scl), f2bf(xv.w * scl) };
    *(uint2*)&out[(size_t)row * D_DIM + c0] = *(uint2*)t;
}

// ---------------------------------------------------------------------------
// Kernel 5: bf16 matrix transpose [M_ROWS, D] -> [D, M_ROWS]
// ---------------------------------------------------------------------------
__global__ void __launch_bounds__(256)
transpose_kernel(const unsigned short* __restrict__ in,   // [M][D]
                 unsigned short* __restrict__ out)        // [D][M]
{
    __shared__ unsigned short tile[64][68];   // pad to dodge bank conflicts
    const int c0 = blockIdx.x * 64;           // D offset
    const int r0 = blockIdx.y * 64;           // M offset
    const int lr = threadIdx.x >> 4;          // 0..15
    const int lc = (threadIdx.x & 15) * 4;    // 0..60

    #pragma unroll
    for (int rep = 0; rep < 4; rep++) {
        const int row = lr + rep * 16;
        uint2 d = *(const uint2*)&in[(size_t)(r0 + row) * D_DIM + c0 + lc];
        const unsigned short* ps = (const unsigned short*)&d;
        #pragma unroll
        for (int i = 0; i < 4; i++) tile[row][lc + i] = ps[i];
    }
    __syncthreads();
    #pragma unroll
    for (int rep = 0; rep < 4; rep++) {
        const int crow = lr + rep * 16;       // column of in == row of out
        unsigned short t[4];
        #pragma unroll
        for (int i = 0; i < 4; i++) t[i] = tile[lc + i][crow];
        *(uint2*)&out[(size_t)(c0 + crow) * M_ROWS + r0 + lc] = *(uint2*)t;
    }
}

// ---------------------------------------------------------------------------
// Kernel 6: attention, grid = (T/256, B*H), block = 256 (8 waves x 32 q-rows)
// LDS-free / barrier-free. Each wave owns TWO q-tiles so every K A-frag and
// V^T B-frag load feeds two WMMAs (18 WMMAs per 256B/lane of operands).
//
// Exact simplifications vs reference:
//  * |scores| <= beta/8 <= 1.25 -> clip(+-50) no-op, exp can't overflow, and
//    softmax is shift-invariant -> max-subtract dropped (exact math).
//  * log2(e) pre-folded into q -> bare v_exp_f32 (no mul, no denorm guard).
//  * denominator l = P @ ones via WMMA (no lane shuffles).
//  * scores computed TRANSPOSED (S^T = K_tile x Q^T): S^T's C-register layout
//    (lane = query, VGPR v = key v+8*half) is exactly the A-operand layout
//    for O += P x V -> exp+pack stays in registers.
// ---------------------------------------------------------------------------
__global__ void __launch_bounds__(256)
attn_kernel(const unsigned short* __restrict__ q,
            const unsigned short* __restrict__ k,
            const unsigned short* __restrict__ vt,   // [D][B*T]
            unsigned short* __restrict__ o)
{
    const int tid  = threadIdx.x;
    const int lane = tid & 31, wid = tid >> 5;
    const int half = lane >> 4, r = lane & 15;

    const int bh = blockIdx.y;
    const int b  = bh >> 4, h = bh & 15;
    const int cb = h * DH_DIM;
    const int qrow0 = b * T_SEQ + blockIdx.x * 256 + wid * 32;

    // Q as B-operand of S^T GEMM: lane -> query col r, d chunks 8h.. and 16+8h..
    FragB qB[2][2];                       // [q-tile][k-step]
    #pragma unroll
    for (int qt = 0; qt < 2; qt++)
        #pragma unroll
        for (int ks = 0; ks < 2; ks++) {
            const unsigned short* qp =
                q + (size_t)(qrow0 + qt * 16 + r) * D_DIM + cb + ks * 32;
            qB[qt][ks].u[0] = *(const uint4*)(qp + half * 8);
            qB[qt][ks].u[1] = *(const uint4*)(qp + 16 + half * 8);
        }

    // constant ones B-operand (bf16 1.0 pairs) for the row-sum WMMA
    FragB ones;
    #pragma unroll
    for (int i = 0; i < 2; i++) {
        ones.u[i].x = 0x3F803F80u; ones.u[i].y = 0x3F803F80u;
        ones.u[i].z = 0x3F803F80u; ones.u[i].w = 0x3F803F80u;
    }

    const v8f vzero = {0.f, 0.f, 0.f, 0.f, 0.f, 0.f, 0.f, 0.f};
    v8f oacc[2][4], lacc[2];
    #pragma unroll
    for (int qt = 0; qt < 2; qt++) {
        lacc[qt] = vzero;
        #pragma unroll
        for (int j = 0; j < 4; j++) oacc[qt][j] = vzero;
    }

    for (int kt = 0; kt < T_SEQ / 32; kt++) {
        const int krow0 = b * T_SEQ + kt * 32;

        if (kt + 1 < T_SEQ / 32) {  // prefetch next tiles (global_prefetch_b8)
            __builtin_prefetch(
                &k[(size_t)(krow0 + 32 + lane) * D_DIM + cb], 0, 1);
            __builtin_prefetch(
                &vt[(size_t)(cb + lane) * M_ROWS + krow0 + 32], 0, 1);
        }

        // ---- S^T = K_tile(2x 16 keys x 64d) x Q^T, for both q-tiles ----
        v8f s00 = vzero, s10 = vzero;     // q-tile 0: keys 0-15 / 16-31
        v8f s01 = vzero, s11 = vzero;     // q-tile 1: keys 0-15 / 16-31
        #pragma unroll
        for (int ks = 0; ks < 2; ks++) {
            FragB ka0, ka1;
            const unsigned short* kp0 =
                k + (size_t)(krow0 + r) * D_DIM + cb + ks * 32;
            const unsigned short* kp1 =
                k + (size_t)(krow0 + 16 + r) * D_DIM + cb + ks * 32;
            ka0.u[0] = *(const uint4*)(kp0 + half * 8);
            ka0.u[1] = *(const uint4*)(kp0 + 16 + half * 8);
            ka1.u[0] = *(const uint4*)(kp1 + half * 8);
            ka1.u[1] = *(const uint4*)(kp1 + 16 + half * 8);
            s00 = __builtin_amdgcn_wmma_f32_16x16x32_bf16(
                      false, ka0.v, false, qB[0][ks].v, (short)0, s00, false, false);
            s10 = __builtin_amdgcn_wmma_f32_16x16x32_bf16(
                      false, ka1.v, false, qB[0][ks].v, (short)0, s10, false, false);
            s01 = __builtin_amdgcn_wmma_f32_16x16x32_bf16(
                      false, ka0.v, false, qB[1][ks].v, (short)0, s01, false, false);
            s11 = __builtin_amdgcn_wmma_f32_16x16x32_bf16(
                      false, ka1.v, false, qB[1][ks].v, (short)0, s11, false, false);
        }

        // ---- exp2 + pack: S^T C-regs become the A-operand of P x V ----
        // pa element i<8 : key = 8*half + i      ; i>=8: key = 16 + 8*half + (i-8)
        FragB pa0, pa1;
        #pragma unroll
        for (int i = 0; i < 8; i++) {
            pa0.s[i]     = f2bf(fast_exp2(s00[i]));
            pa0.s[8 + i] = f2bf(fast_exp2(s10[i]));
            pa1.s[i]     = f2bf(fast_exp2(s01[i]));
            pa1.s[8 + i] = f2bf(fast_exp2(s11[i]));
        }

        // ---- row-sums (denominators) and O += P x V ----
        lacc[0] = __builtin_amdgcn_wmma_f32_16x16x32_bf16(
                      false, pa0.v, false, ones.v, (short)0, lacc[0], false, false);
        lacc[1] = __builtin_amdgcn_wmma_f32_16x16x32_bf16(
                      false, pa1.v, false, ones.v, (short)0, lacc[1], false, false);
        #pragma unroll
        for (int j = 0; j < 4; j++) {
            FragB bv;
            const unsigned short* vp =
                vt + (size_t)(cb + j * 16 + r) * M_ROWS + krow0;
            bv.u[0] = *(const uint4*)(vp + half * 8);
            bv.u[1] = *(const uint4*)(vp + 16 + half * 8);
            oacc[0][j] = __builtin_amdgcn_wmma_f32_16x16x32_bf16(
                             false, pa0.v, false, bv.v, (short)0, oacc[0][j],
                             false, false);
            oacc[1][j] = __builtin_amdgcn_wmma_f32_16x16x32_bf16(
                             false, pa1.v, false, bv.v, (short)0, oacc[1][j],
                             false, false);
        }
    }

    // epilogue: O / l, store bf16 into [B*T, D] at this head's columns
    #pragma unroll
    for (int qt = 0; qt < 2; qt++)
        #pragma unroll
        for (int t = 0; t < 8; t++) {
            const float inv = 1.0f / lacc[qt][t];
            const int row = qrow0 + qt * 16 + t + 8 * half;
            #pragma unroll
            for (int j = 0; j < 4; j++)
                o[(size_t)row * D_DIM + cb + j * 16 + r] =
                    f2bf(oacc[qt][j][t] * inv);
        }
}

// ---------------------------------------------------------------------------
// Host-side orchestration
// ---------------------------------------------------------------------------
extern "C" void kernel_launch(void* const* d_in, const int* in_sizes, int n_in,
                              void* d_out, int out_size, void* d_ws, size_t ws_size,
                              hipStream_t stream)
{
    (void)in_sizes; (void)n_in; (void)out_size; (void)ws_size;

    const float* query  = (const float*)d_in[0];
    const float* q_w    = (const float*)d_in[1];
    const float* k_w    = (const float*)d_in[2];
    const float* v_w    = (const float*)d_in[3];
    const float* o_w    = (const float*)d_in[4];
    const float* q_b    = (const float*)d_in[5];
    const float* k_b    = (const float*)d_in[6];
    const float* v_b    = (const float*)d_in[7];
    const float* o_b    = (const float*)d_in[8];
    const float* ln_q_g = (const float*)d_in[9];
    const float* ln_q_b = (const float*)d_in[10];
    const float* ln_k_g = (const float*)d_in[11];
    const float* ln_k_b = (const float*)d_in[12];
    const float* ln_v_g = (const float*)d_in[13];
    const float* ln_v_b = (const float*)d_in[14];
    const float* beta_p = (const float*)d_in[15];

    char* ws = (char*)d_ws;
    size_t off = 0;
    auto alloc = [&](size_t bytes) -> char* {
        char* p = ws + off;
        off += (bytes + 255) & ~(size_t)255;
        return p;
    };

    const size_t actB = (size_t)M_ROWS * D_DIM;   // 4M elements
    const size_t wB   = (size_t)D_DIM * D_DIM;    // 1M elements

    unsigned short* qn  = (unsigned short*)alloc(actB * 2);
    unsigned short* kn  = (unsigned short*)alloc(actB * 2);
    unsigned short* vn  = (unsigned short*)alloc(actB * 2);
    unsigned short* wqb = (unsigned short*)alloc(wB * 2);
    unsigned short* wkb = (unsigned short*)alloc(wB * 2);
    unsigned short* wvb = (unsigned short*)alloc(wB * 2);
    unsigned short* wob = (unsigned short*)alloc(wB * 2);
    float*          qf  = (float*)alloc(actB * 4);
    float*          kf  = (float*)alloc(actB * 4);
    unsigned short* vb  = (unsigned short*)alloc(actB * 2);
    unsigned short* vtb = (unsigned short*)alloc(actB * 2);   // V transposed
    // region reuse (stream-serialized): qn/kn/vn fully consumed before rewrite
    unsigned short* qhat = qn;
    unsigned short* khat = kn;
    unsigned short* att  = vn;

    // 1) fused layernorm x3
    ln3_kernel<<<M_ROWS, 256, 0, stream>>>(query,
        ln_q_g, ln_q_b, ln_k_g, ln_k_b, ln_v_g, ln_v_b, qn, kn, vn);

    // 2) weights fp32 -> bf16
    cvt_kernel<<<256, 256, 0, stream>>>(q_w, wqb, (int)wB);
    cvt_kernel<<<256, 256, 0, stream>>>(k_w, wkb, (int)wB);
    cvt_kernel<<<256, 256, 0, stream>>>(v_w, wvb, (int)wB);
    cvt_kernel<<<256, 256, 0, stream>>>(o_w, wob, (int)wB);

    // 3) projections (WMMA bf16 -> f32)
    dim3 ggrid(M_ROWS / 64, D_DIM / 128);
    gemm_nt<0, 1, 0><<<ggrid, 256, 0, stream>>>(qn, wqb, q_b, nullptr, qf,
                                                M_ROWS, D_DIM, D_DIM);
    gemm_nt<0, 1, 0><<<ggrid, 256, 0, stream>>>(kn, wkb, k_b, nullptr, kf,
                                                M_ROWS, D_DIM, D_DIM);
    gemm_nt<1, 1, 0><<<ggrid, 256, 0, stream>>>(vn, wvb, v_b, nullptr, vb,
                                                M_ROWS, D_DIM, D_DIM);

    // 4) l2norm (+ beta/8*log2e folded into q); V pre-transpose for attention
    l2norm_kernel<<<M_ROWS, 256, 0, stream>>>(qf, qhat, beta_p, 1);
    l2norm_kernel<<<M_ROWS, 256, 0, stream>>>(kf, khat, beta_p, 0);
    transpose_kernel<<<dim3(D_DIM / 64, M_ROWS / 64), 256, 0, stream>>>(vb, vtb);

    // 5) attention (WMMA-dense, LDS-free, shuffle-free softmax)
    attn_kernel<<<dim3(T_SEQ / 256, B_BATCH * H_HEADS), 256, 0, stream>>>(
        qhat, khat, vtb, att);

    // 6) output projection + bias + residual -> fp32 d_out
    gemm_nt<0, 1, 1><<<ggrid, 256, 0, stream>>>(att, wob, o_b, query,
                                                (float*)d_out,
                                                M_ROWS, D_DIM, D_DIM);
}
